// ScaledDotProductAttention_55121610277486
// MI455X (gfx1250) — matrix operand order repaired and checked by reading, not compiled, not run on hardware
//
#include <hip/hip_runtime.h>
#include <hip/hip_bf16.h>
#include <math.h>
#include <stdint.h>

typedef __attribute__((ext_vector_type(16))) _Float16 v16h;
typedef __attribute__((ext_vector_type(8)))  float    v8f;
typedef __attribute__((ext_vector_type(4)))  unsigned int u32x4;
typedef __attribute__((ext_vector_type(8)))  int      i32x8;
typedef __attribute__((ext_vector_type(4)))  int      i32x4;

namespace {
constexpr int kB = 2, kH = 12, kS = 2048, kD = 64;
constexpr size_t kCtxElems = (size_t)kB * kH * kS * kD;   // 3,145,728
}

// Pack 8 fp32 values (two float4) into halves [base, base+8) of an A/B fragment.
__device__ __forceinline__ void pack8(v16h& dst, int base, float4 lo, float4 hi) {
  dst[base + 0] = (_Float16)lo.x; dst[base + 1] = (_Float16)lo.y;
  dst[base + 2] = (_Float16)lo.z; dst[base + 3] = (_Float16)lo.w;
  dst[base + 4] = (_Float16)hi.x; dst[base + 5] = (_Float16)hi.y;
  dst[base + 6] = (_Float16)hi.z; dst[base + 7] = (_Float16)hi.w;
}

// T5 bidirectional relative-position bucket bias. num_buckets=32 -> 16 per side,
// max_exact=8, max_distance=128. log(n/8)/log(128/8)*8 == 2*log2(n) - 6.
__device__ __forceinline__ float t5_bias(const float* __restrict__ rel_bias,
                                         int h, int q, int k) {
  int n = q - k;                       // n = -(k - q)
  int ret = (n < 0) ? 16 : 0;
  n = (n < 0) ? -n : n;
  int bucket;
  if (n < 8) {
    bucket = n;
  } else {
    int v = 8 + (int)(2.0f * __log2f((float)n) - 6.0f);   // trunc toward zero
    bucket = (v < 15) ? v : 15;
  }
  return rel_bias[(ret + bucket) * kH + h];
}

// ---------------------------------------------------------------------------
// Tensor Data Mover: 1-D copy of 2048 consecutive f32 (8 KB) global -> LDS.
// D# built per CDNA5 ISA.
// ---------------------------------------------------------------------------
__device__ __forceinline__ void tdm_load_chunk(const float* gsrc, unsigned lds_off) {
  unsigned long long ga = (unsigned long long)(uintptr_t)gsrc;
  u32x4 g0;
  g0[0] = 1u;                                   // count=1, is_restore=0, no gather
  g0[1] = lds_off;                              // lds_addr (bytes)
  g0[2] = (unsigned)(ga & 0xFFFFFFFFu);         // global_addr[31:0]
  g0[3] = (unsigned)((ga >> 32) & 0x01FFFFFFu)  // global_addr[56:32]
        | (2u << 30);                           // type=2 ("image")
  i32x8 g1;
  g1[0] = 0x00020000;                           // data_size=2 (4B); mask=0
  g1[1] = (int)((2048u & 0xFFFFu) << 16);       // tensor_dim0[15:0]=2048 @bits63:48
  g1[2] = (int)(1u << 16);                      // tensor_dim0 hi=0; tensor_dim1=1
  g1[3] = (int)(2048u << 16);                   // tensor_dim1 hi=0; tile_dim0=2048
  g1[4] = 1;                                    // tile_dim1=1; tile_dim2=0
  g1[5] = 2048;                                 // tensor_dim0_stride[31:0]
  g1[6] = 0;                                    // stride hi / tensor_dim1_stride lo
  g1[7] = 0;                                    // tensor_dim1_stride hi
  i32x4 z4 = {0, 0, 0, 0};                      // dims >= 2 unused
  i32x8 z8 = {0, 0, 0, 0, 0, 0, 0, 0};          // unused trailing group (6-arg form)
  __builtin_amdgcn_tensor_load_to_lds(g0, g1, z4, z4, z8, 0);
}

// ---------------------------------------------------------------------------
// Pass 1: logits[b,h,q,k] = (Q.K^T)*scale + bias(q,k,h), masked.
// 4 waves / WG; wave w computes the 16x16 tile at (q0, k0 + 16*w).
// ---------------------------------------------------------------------------
__global__ __launch_bounds__(128) void qk_logits_kernel(
    const float* __restrict__ Q, const float* __restrict__ K,
    const unsigned char* __restrict__ mask, const float* __restrict__ rel_bias,
    float* __restrict__ prob) {
  const int lane = threadIdx.x & 31;
  const int wave = threadIdx.x >> 5;
  const int m    = lane & 15;          // row (A) / column (B,C) selector
  const int hs   = lane >> 4;          // half-wave selector
  const int bh   = blockIdx.z;
  const int b    = bh / kH;
  const int h    = bh - b * kH;
  const int q0   = blockIdx.x * 16;
  const int k0   = blockIdx.y * 64 + wave * 16;

  const float* qrow = Q + ((size_t)bh * kS + q0 + m) * kD;   // A: row q0+m
  const float* krow = K + ((size_t)bh * kS + k0 + m) * kD;   // B: column k0+m

  v8f c = {};
#pragma unroll
  for (int dc = 0; dc < kD; dc += 32) {
    v16h a, bm;
    // A fragment (16x32 f16): lane half hs -> K {hs*8..hs*8+7, 16+hs*8..}
    float4 a0 = *(const float4*)(qrow + dc + hs * 8);
    float4 a1 = *(const float4*)(qrow + dc + hs * 8 + 4);
    float4 a2 = *(const float4*)(qrow + dc + 16 + hs * 8);
    float4 a3 = *(const float4*)(qrow + dc + 16 + hs * 8 + 4);
    pack8(a, 0, a0, a1);
    pack8(a, 8, a2, a3);
    // B fragment (32x16 f16): lane half hs -> K {hs*8..hs*8+7, 16+hs*8..}, column = k0+m
    float4 b0 = *(const float4*)(krow + dc + hs * 8);
    float4 b1 = *(const float4*)(krow + dc + hs * 8 + 4);
    float4 b2 = *(const float4*)(krow + dc + 16 + hs * 8);
    float4 b3 = *(const float4*)(krow + dc + 16 + hs * 8 + 4);
    pack8(bm, 0, b0, b1);
    pack8(bm, 8, b2, b3);
    c = __builtin_amdgcn_wmma_f32_16x16x32_f16(false, a, false, bm,
                                               (short)0, c, false, false);
  }

  const float scale = 0.125f;          // 1/sqrt(64)
  const unsigned char mk = mask[b * kS + k0 + m];   // mask broadcast over q
  float* prow = prob + ((size_t)bh * kS + q0) * kS + k0;
#pragma unroll
  for (int i = 0; i < 8; ++i) {
    const int row = hs * 8 + i;        // C/D layout: M = i + 8*hs, N = lane&15
    float s = c[i] * scale + t5_bias(rel_bias, h, q0 + row, k0 + m);
    if (mk) s = -1e9f;
    prow[(size_t)row * kS + m] = s;
  }
}

// ---------------------------------------------------------------------------
// Pass 2: in-place row softmax over 2048 elements. One WG per row.
// ---------------------------------------------------------------------------
__global__ __launch_bounds__(256) void softmax_rows_kernel(float* __restrict__ prob) {
  __shared__ float red[256];
  float* p = prob + (size_t)blockIdx.x * kS;
  const int t = threadIdx.x;

  float v[8];
  float m = -INFINITY;
#pragma unroll
  for (int i = 0; i < 8; ++i) {
    v[i] = p[t + i * 256];
    m = fmaxf(m, v[i]);
  }
  red[t] = m;
  __syncthreads();
  for (int s = 128; s > 0; s >>= 1) {
    if (t < s) red[t] = fmaxf(red[t], red[t + s]);
    __syncthreads();
  }
  m = red[0];
  __syncthreads();

  float sum = 0.0f;
#pragma unroll
  for (int i = 0; i < 8; ++i) {
    v[i] = __expf(v[i] - m);
    sum += v[i];
  }
  red[t] = sum;
  __syncthreads();
  for (int s = 128; s > 0; s >>= 1) {
    if (t < s) red[t] += red[t + s];
    __syncthreads();
  }
  const float inv = 1.0f / red[0];
#pragma unroll
  for (int i = 0; i < 8; ++i) p[t + i * 256] = v[i] * inv;
}

// ---------------------------------------------------------------------------
// Pass 3: context = P x V. 4 waves / WG; wave w owns D slice n0 = 16*w.
// V chunks (32 x 64 f32 = 8 KB, contiguous) are DMA'd into a double-buffered
// LDS region by the Tensor Data Mover while all waves run WMMA on the
// previous chunk. Wave 0 issues the TDM op and waits on TENSORcnt before the
// workgroup barrier publishes the buffer.
// ---------------------------------------------------------------------------
__global__ __launch_bounds__(128) void pv_context_kernel(
    const float* __restrict__ prob, const float* __restrict__ V,
    float* __restrict__ ctx) {
  __shared__ float vbuf[2][32 * kD];   // 2 x 8 KB double buffer
  const int lane = threadIdx.x & 31;
  const int wave = threadIdx.x >> 5;
  const int m    = lane & 15;
  const int hs   = lane >> 4;
  const int bh   = blockIdx.z;
  const int q0   = blockIdx.x * 16;
  const int n0   = wave * 16;

  const float* prow  = prob + ((size_t)bh * kS + q0 + m) * kS;
  const float* vbase = V + (size_t)bh * kS * kD;
  const unsigned lds0 = (unsigned)(uintptr_t)&vbuf[0][0];
  const unsigned lds1 = (unsigned)(uintptr_t)&vbuf[1][0];

  // Prologue: DMA chunk 0 into buffer 0.
  if (wave == 0) {
    tdm_load_chunk(vbase, lds0);
    __builtin_amdgcn_s_wait_tensorcnt(0);
  }
  __syncthreads();

  v8f c = {};
  for (int kc = 0; kc < kS; kc += 32) {
    const int cur = (kc >> 5) & 1;

    // Kick off DMA of the next chunk into the other buffer (overlaps WMMA).
    if (wave == 0 && kc + 32 < kS)
      tdm_load_chunk(vbase + (size_t)(kc + 32) * kD, cur ? lds0 : lds1);

    __builtin_prefetch(prow + kc + 128, 0, 0);   // global_prefetch_b8 on P stream

    // A fragment from P (f32 global -> f16).
    v16h a, bm;
    float4 p0 = *(const float4*)(prow + kc + hs * 8);
    float4 p1 = *(const float4*)(prow + kc + hs * 8 + 4);
    float4 p2 = *(const float4*)(prow + kc + 16 + hs * 8);
    float4 p3 = *(const float4*)(prow + kc + 16 + hs * 8 + 4);
    pack8(a, 0, p0, p1);
    pack8(a, 8, p2, p3);

    // B fragment from the LDS V chunk: element (K = j<8 ? hs*8+j : 16+hs*8+j-8, N = n0+m).
    const float* vb = &vbuf[cur][0];
    const int n = n0 + m;
#pragma unroll
    for (int j = 0; j < 16; ++j) bm[j] = (_Float16)vb[((j < 8) ? (hs * 8 + j) : (16 + hs * 8 + (j - 8))) * kD + n];

    c = __builtin_amdgcn_wmma_f32_16x16x32_f16(false, a, false, bm,
                                               (short)0, c, false, false);

    // Publish the next buffer: issuing wave drains TENSORcnt, then barrier.
    if (wave == 0 && kc + 32 < kS)
      __builtin_amdgcn_s_wait_tensorcnt(0);
    __syncthreads();
  }

  float* crow = ctx + ((size_t)bh * kS + q0) * kD + n0;
#pragma unroll
  for (int i = 0; i < 8; ++i) {
    const int row = hs * 8 + i;
    crow[(size_t)row * kD + m] = c[i];
  }
}

extern "C" void kernel_launch(void* const* d_in, const int* in_sizes, int n_in,
                              void* d_out, int out_size, void* d_ws, size_t ws_size,
                              hipStream_t stream) {
  const float* Q = (const float*)d_in[0];
  const float* K = (const float*)d_in[1];
  const float* V = (const float*)d_in[2];
  const unsigned char* mask = (const unsigned char*)d_in[3];
  const float* rel_bias = (const float*)d_in[4];

  float* ctx  = (float*)d_out;               // output 0: context [B,H,S,D]
  float* prob = (float*)d_out + kCtxElems;   // output 1: attn_prob [B,H,S,S]

  dim3 g1(kS / 16, kS / 64, kB * kH);        // (128, 32, 24)
  qk_logits_kernel<<<g1, 128, 0, stream>>>(Q, K, mask, rel_bias, prob);

  softmax_rows_kernel<<<dim3(kB * kH * kS), 256, 0, stream>>>(prob);

  dim3 g3(kS / 16, 1, kB * kH);              // (128, 1, 24)
  pv_context_kernel<<<g3, 128, 0, stream>>>(prob, V, ctx);
}
